// MultiHeadCrossModalAttention_53953379172737
// MI455X (gfx1250) — compile-verified, matrix operand order checked by reading, not run on hardware
//
#include <hip/hip_runtime.h>
#include <hip/hip_bf16.h>

// ---------------------------------------------------------------------------
// MultiHeadCrossModalAttention on MI455X (gfx1250, wave32, WMMA bf16,
// async global->LDS staging, DPP16 softmax reductions)
// B=4, S=2048, D=512, H=8, HD=64
// ---------------------------------------------------------------------------

typedef __attribute__((ext_vector_type(16))) __bf16 v16bf;
typedef __attribute__((ext_vector_type(8)))  __bf16 v8bf;
typedef __attribute__((ext_vector_type(8)))  float  v8f;
typedef int v4i __attribute__((vector_size(16)));

#define B_  4
#define S_  2048
#define D_  512
#define H_  8
#define HD_ 64
#define MROWS (B_ * S_)          // 8192

#define AS1 __attribute__((address_space(1)))
#define AS3 __attribute__((address_space(3)))

#if __has_builtin(__builtin_amdgcn_global_load_async_to_lds_b128) && \
    __has_builtin(__builtin_amdgcn_s_wait_asynccnt)
#define HAVE_ASYNC_LDS 1
#else
#define HAVE_ASYNC_LDS 0
#endif

#if __has_builtin(__builtin_amdgcn_update_dpp)
#define HAVE_DPP 1
#else
#define HAVE_DPP 0
#endif

// one 16B chunk: global -> LDS (async on CDNA5, sync fallback otherwise)
__device__ __forceinline__ void stage16(const __bf16* g, __bf16* l) {
#if HAVE_ASYNC_LDS
  __builtin_amdgcn_global_load_async_to_lds_b128((AS1 v4i*)g, (AS3 v4i*)l,
                                                 0, 0);
#else
  *(v8bf*)l = *(const v8bf*)g;
#endif
}

__device__ __forceinline__ void async_wait_all() {
#if HAVE_ASYNC_LDS
  __builtin_amdgcn_s_wait_asynccnt(0);
#endif
}

#if HAVE_DPP
template <int CTRL>
__device__ __forceinline__ float dpp_mov_f32(float x) {
  int r = __builtin_amdgcn_update_dpp(0, __float_as_int(x), CTRL, 0xF, 0xF,
                                      true);
  return __int_as_float(r);
}
#endif

// reduce (max / sum) across each 16-lane group; all lanes get the result.
__device__ __forceinline__ float red16_max(float x) {
#if HAVE_DPP
  x = fmaxf(x, dpp_mov_f32<0xB1>(x));    // quad_perm(1,0,3,2): xor 1
  x = fmaxf(x, dpp_mov_f32<0x4E>(x));    // quad_perm(2,3,0,1): xor 2
  x = fmaxf(x, dpp_mov_f32<0x141>(x));   // row_half_mirror: cross-quad in 8
  x = fmaxf(x, dpp_mov_f32<0x140>(x));   // row_mirror: cross-8 in 16
#else
#pragma unroll
  for (int off = 1; off < 16; off <<= 1) x = fmaxf(x, __shfl_xor(x, off, 32));
#endif
  return x;
}

__device__ __forceinline__ float red16_sum(float x) {
#if HAVE_DPP
  x += dpp_mov_f32<0xB1>(x);
  x += dpp_mov_f32<0x4E>(x);
  x += dpp_mov_f32<0x141>(x);
  x += dpp_mov_f32<0x140>(x);
#else
#pragma unroll
  for (int off = 1; off < 16; off <<= 1) x += __shfl_xor(x, off, 32);
#endif
  return x;
}

__device__ __forceinline__ v16bf pack16(v8bf lo, v8bf hi) {
  union { v16bf v; v8bf p[2]; } u;
  u.p[0] = lo; u.p[1] = hi;
  return u.v;
}

__device__ __forceinline__ v8f zero8() {
  v8f z;
#pragma unroll
  for (int i = 0; i < 8; ++i) z[i] = 0.0f;
  return z;
}

// ------------------------------- casts -------------------------------------

__global__ __launch_bounds__(256) void cast_bf16_kernel(
    const float* __restrict__ in, __bf16* __restrict__ out, int n) {
  int i = (blockIdx.x * 256 + threadIdx.x) * 4;
  if (i + 3 < n) {
#pragma unroll
    for (int j = 0; j < 4; ++j) out[i + j] = (__bf16)in[i + j];
  }
}

// Wt[n][k] = W[k][n], 512x512
__global__ __launch_bounds__(256) void castT_bf16_kernel(
    const float* __restrict__ W, __bf16* __restrict__ Wt) {
  int i = blockIdx.x * 256 + threadIdx.x;   // over 512*512
  int n = i >> 9;
  int k = i & 511;
  Wt[i] = (__bf16)W[k * D_ + n];
}

// ------------------------------- GEMM --------------------------------------
// C[M,N] = (A[M,K] @ Bt[N,K]^T + bias[N]) * alpha
// Tiles: BM=128, BN=128, BK=32.  256 threads = 8 waves (2x4), wave: 64x32.
// Double-buffered async global->LDS staging.

#define BM 128
#define BN 128
#define BK 32

template <bool OUT_F32>
__global__ __launch_bounds__(256) void gemm_bf16_kernel(
    const __bf16* __restrict__ A, const __bf16* __restrict__ Bt,
    const float* __restrict__ bias, void* __restrict__ C,
    int M, int N, int K, float alpha) {
  __shared__ __align__(16) __bf16 As[2][BM][BK + 8];   // 80B rows
  __shared__ __align__(16) __bf16 Bs[2][BM][BK + 8];

  const int tid  = threadIdx.x;
  const int bm0  = blockIdx.y * BM;
  const int bn0  = blockIdx.x * BN;
  const int wave = tid >> 5;
  const int lane = tid & 31;
  const int wm   = wave >> 2;        // 0..1 -> 64-row slab
  const int wn   = wave & 3;         // 0..3 -> 32-col slab
  const int lr   = lane & 15;
  const int lh   = lane >> 4;
  const int koff = lh * 8;

  // this thread's two 16B staging chunks (same mapping for A and B tiles)
  const int c0   = tid * 2;
  const int row0 = c0 >> 2;
  const int col0 = (c0 & 3) * 8;
  const int row1 = (c0 + 1) >> 2;
  const int col1 = ((c0 + 1) & 3) * 8;

  v8f acc[4][2];
#pragma unroll
  for (int mt = 0; mt < 4; ++mt)
#pragma unroll
    for (int nt = 0; nt < 2; ++nt) acc[mt][nt] = zero8();

  // prologue: stage tile 0 into buffer 0
  {
    stage16(&A[(size_t)(bm0 + row0) * K + col0], &As[0][row0][col0]);
    stage16(&Bt[(size_t)(bn0 + row0) * K + col0], &Bs[0][row0][col0]);
    stage16(&A[(size_t)(bm0 + row1) * K + col1], &As[0][row1][col1]);
    stage16(&Bt[(size_t)(bn0 + row1) * K + col1], &Bs[0][row1][col1]);
  }

  int p = 0;
  for (int bk = 0; bk < K; bk += BK, p ^= 1) {
    async_wait_all();
    __syncthreads();
    if (bk + BK < K) {
      int nb = bk + BK;
      stage16(&A[(size_t)(bm0 + row0) * K + nb + col0], &As[p ^ 1][row0][col0]);
      stage16(&Bt[(size_t)(bn0 + row0) * K + nb + col0], &Bs[p ^ 1][row0][col0]);
      stage16(&A[(size_t)(bm0 + row1) * K + nb + col1], &As[p ^ 1][row1][col1]);
      stage16(&Bt[(size_t)(bn0 + row1) * K + nb + col1], &Bs[p ^ 1][row1][col1]);
    }

    v16bf af[4], bf[2];
#pragma unroll
    for (int mt = 0; mt < 4; ++mt) {
      int r   = wm * 64 + mt * 16 + lr;
      v8bf lo = *(const v8bf*)&As[p][r][koff];
      v8bf hi = *(const v8bf*)&As[p][r][koff + 16];
      af[mt]  = pack16(lo, hi);
    }
#pragma unroll
    for (int nt = 0; nt < 2; ++nt) {
      int r   = wn * 32 + nt * 16 + lr;
      v8bf lo = *(const v8bf*)&Bs[p][r][koff];
      v8bf hi = *(const v8bf*)&Bs[p][r][koff + 16];
      bf[nt]  = pack16(lo, hi);
    }
#pragma unroll
    for (int mt = 0; mt < 4; ++mt)
#pragma unroll
      for (int nt = 0; nt < 2; ++nt)
        acc[mt][nt] = __builtin_amdgcn_wmma_f32_16x16x32_bf16(
            false, af[mt], false, bf[nt], (short)0, acc[mt][nt], false, false);
  }

  // epilogue: C layout -> lane holds col n=lr of each tile, rows r + lh*8
#pragma unroll
  for (int nt = 0; nt < 2; ++nt) {
    int n    = bn0 + wn * 32 + nt * 16 + lr;
    float bv = bias[n];
#pragma unroll
    for (int mt = 0; mt < 4; ++mt) {
#pragma unroll
      for (int r = 0; r < 8; ++r) {
        int m   = bm0 + wm * 64 + mt * 16 + lh * 8 + r;
        float v = (acc[mt][nt][r] + bv) * alpha;
        if constexpr (OUT_F32)
          ((float*)C)[(size_t)m * N + n] = v;
        else
          ((__bf16*)C)[(size_t)m * N + n] = (__bf16)v;
      }
    }
  }
}

// ------------------------------- fuse --------------------------------------
// K' = K*s + sh ; V' = V*s + sh ; also write V' transposed per head:
// Vt[b][h][hd][s]
__global__ __launch_bounds__(256) void fuse_kv_kernel(
    const __bf16* __restrict__ Kr, const __bf16* __restrict__ Vr,
    const __bf16* __restrict__ Sc, const __bf16* __restrict__ Sh,
    __bf16* __restrict__ Kf, __bf16* __restrict__ Vt) {
  int i = blockIdx.x * 256 + threadIdx.x;      // over B*S*D = 2^22
  float sc = (float)Sc[i], sh = (float)Sh[i];
  float kv = (float)Kr[i] * sc + sh;
  float vv = (float)Vr[i] * sc + sh;
  Kf[i] = (__bf16)kv;
  int d = i & (D_ - 1);
  int s = (i >> 9) & (S_ - 1);
  int b = i >> 20;
  int h = d >> 6, hd = d & 63;
  Vt[(((size_t)(b * H_ + h) * HD_) + hd) * S_ + s] = (__bf16)vv;
}

// ---------------------------- flash attention -------------------------------
// grid: (S/64, H, B), block 128 (4 waves). Wave w owns q rows q0+w*16..+15.
// Q already scaled by 1/sqrt(HD). Double-buffered async K/V tile staging.
__global__ __launch_bounds__(128) void attn_kernel(
    const __bf16* __restrict__ Qb, const __bf16* __restrict__ Kb,
    const __bf16* __restrict__ Vt, __bf16* __restrict__ Ob) {
  __shared__ __align__(16) __bf16 Ks[2][64][72];    // [key][hd]
  __shared__ __align__(16) __bf16 Vs[2][64][72];    // [hd][key] (transposed V)
  __shared__ __align__(16) __bf16 Ps[4][16][72];    // per-wave P tile

  const int b    = blockIdx.z;
  const int h    = blockIdx.y;
  const int q0   = blockIdx.x * 64;
  const int tid  = threadIdx.x;
  const int wave = tid >> 5;
  const int lane = tid & 31;
  const int lr   = lane & 15;
  const int lh   = lane >> 4;
  const int koff = lh * 8;

  const __bf16* Kbase = Kb + (size_t)(b * S_) * D_ + h * HD_;
  const __bf16* Vbase = Vt + (size_t)(b * H_ + h) * HD_ * S_;

  // Q fragments (16x64 per wave -> 2 k-steps), resident in VGPRs
  v16bf qf[2];
  {
    int qrow           = q0 + wave * 16 + lr;
    const __bf16* qptr = Qb + (size_t)(b * S_ + qrow) * D_ + h * HD_;
#pragma unroll
    for (int ks = 0; ks < 2; ++ks) {
      v8bf lo = *(const v8bf*)(qptr + ks * 32 + koff);
      v8bf hi = *(const v8bf*)(qptr + ks * 32 + koff + 16);
      qf[ks]  = pack16(lo, hi);
    }
  }

  v8f o[4];
#pragma unroll
  for (int nt = 0; nt < 4; ++nt) o[nt] = zero8();
  float rm[8], rs[8];
#pragma unroll
  for (int r = 0; r < 8; ++r) { rm[r] = -1e30f; rs[r] = 0.0f; }

  // this thread's four 16B staging chunks per tile
  // chunk c = tid + i*128; row = c>>3 (8 x 16B per 128B row); col = (c&7)*8
  // prologue: stage tile 0 into buffer 0
#pragma unroll
  for (int i = 0; i < 4; ++i) {
    int c = tid + i * 128, row = c >> 3, col = (c & 7) * 8;
    stage16(&Kbase[(size_t)row * D_ + col], &Ks[0][row][col]);
    stage16(&Vbase[(size_t)row * S_ + col], &Vs[0][row][col]);
  }

  int p = 0;
  for (int kt0 = 0; kt0 < S_; kt0 += 64, p ^= 1) {
    async_wait_all();
    __syncthreads();
    if (kt0 + 64 < S_) {
      int nk = kt0 + 64;
#pragma unroll
      for (int i = 0; i < 4; ++i) {
        int c = tid + i * 128, row = c >> 3, col = (c & 7) * 8;
        stage16(&Kbase[(size_t)(nk + row) * D_ + col], &Ks[p ^ 1][row][col]);
        stage16(&Vbase[(size_t)row * S_ + nk + col], &Vs[p ^ 1][row][col]);
      }
    }

    // S = Q @ K^T   (4 n-tiles x 2 k-steps)
    v8f sc[4];
#pragma unroll
    for (int nt = 0; nt < 4; ++nt) {
      sc[nt] = zero8();
#pragma unroll
      for (int ks = 0; ks < 2; ++ks) {
        int row  = nt * 16 + lr;
        v8bf lo  = *(const v8bf*)&Ks[p][row][ks * 32 + koff];
        v8bf hi  = *(const v8bf*)&Ks[p][row][ks * 32 + koff + 16];
        v16bf kf = pack16(lo, hi);
        sc[nt]   = __builtin_amdgcn_wmma_f32_16x16x32_bf16(
            false, qf[ks], false, kf, (short)0, sc[nt], false, false);
      }
    }

    // online softmax: lane holds col n=lr, rows r+lh*8; row lives in 16 lanes
#pragma unroll
    for (int r = 0; r < 8; ++r) {
      float mx = fmaxf(fmaxf(sc[0][r], sc[1][r]), fmaxf(sc[2][r], sc[3][r]));
      mx = red16_max(mx);
      float nm   = fmaxf(rm[r], mx);
      float corr = __expf(rm[r] - nm);
      rm[r]      = nm;
      float psum = 0.0f;
#pragma unroll
      for (int nt = 0; nt < 4; ++nt) {
        float pv  = __expf(sc[nt][r] - nm);
        sc[nt][r] = pv;
        psum += pv;
      }
      psum  = red16_sum(psum);
      rs[r] = rs[r] * corr + psum;
#pragma unroll
      for (int nt = 0; nt < 4; ++nt) o[nt][r] *= corr;
    }

    // transpose P through LDS into A-fragment layout (per-wave region;
    // same-wave LDS ops are in-order, no block barrier needed)
#pragma unroll
    for (int nt = 0; nt < 4; ++nt)
#pragma unroll
      for (int r = 0; r < 8; ++r)
        Ps[wave][lh * 8 + r][nt * 16 + lr] = (__bf16)sc[nt][r];

    // O += P @ V   (2 k-steps over keys x 4 hd n-tiles)
#pragma unroll
    for (int ks = 0; ks < 2; ++ks) {
      v8bf plo = *(const v8bf*)&Ps[wave][lr][ks * 32 + koff];
      v8bf phi = *(const v8bf*)&Ps[wave][lr][ks * 32 + koff + 16];
      v16bf pf = pack16(plo, phi);
#pragma unroll
      for (int nt = 0; nt < 4; ++nt) {
        int row  = nt * 16 + lr;
        v8bf lo  = *(const v8bf*)&Vs[p][row][ks * 32 + koff];
        v8bf hi  = *(const v8bf*)&Vs[p][row][ks * 32 + koff + 16];
        v16bf vf = pack16(lo, hi);
        o[nt]    = __builtin_amdgcn_wmma_f32_16x16x32_bf16(
            false, pf, false, vf, (short)0, o[nt], false, false);
      }
    }
  }

  // normalize and store (bf16, [B*S, D] layout)
#pragma unroll
  for (int r = 0; r < 8; ++r) {
    float inv = 1.0f / rs[r];
    int m     = q0 + wave * 16 + lh * 8 + r;
#pragma unroll
    for (int nt = 0; nt < 4; ++nt)
      Ob[(size_t)(b * S_ + m) * D_ + h * HD_ + nt * 16 + lr] =
          (__bf16)(o[nt][r] * inv);
  }
}

// ------------------------------ launcher ------------------------------------

extern "C" void kernel_launch(void* const* d_in, const int* in_sizes, int n_in,
                              void* d_out, int out_size, void* d_ws,
                              size_t ws_size, hipStream_t stream) {
  const float* mod1 = (const float*)d_in[0];
  const float* mod2 = (const float*)d_in[1];
  const float* Wq   = (const float*)d_in[2];
  const float* bq   = (const float*)d_in[3];
  const float* Wk   = (const float*)d_in[4];
  const float* bk   = (const float*)d_in[5];
  const float* Wv   = (const float*)d_in[6];
  const float* bv   = (const float*)d_in[7];
  const float* Wo   = (const float*)d_in[8];
  const float* bo   = (const float*)d_in[9];
  const float* Wsc  = (const float*)d_in[10];
  const float* bsc  = (const float*)d_in[11];
  const float* Wsh  = (const float*)d_in[12];
  const float* bsh  = (const float*)d_in[13];
  float* out        = (float*)d_out;

  const size_t ACT = (size_t)MROWS * D_ * sizeof(__bf16);  // 8 MB
  const size_t WT  = (size_t)D_ * D_ * sizeof(__bf16);     // 0.5 MB

  char* w = (char*)d_ws;
  size_t off = 0;
  auto carve = [&](size_t bytes) -> __bf16* {
    __bf16* p = (__bf16*)(w + off);
    off += bytes;
    return p;
  };
  __bf16* mod1b = carve(ACT);
  __bf16* mod2b = carve(ACT);
  __bf16* Wqt   = carve(WT);
  __bf16* Wkt   = carve(WT);
  __bf16* Wvt   = carve(WT);
  __bf16* Wot   = carve(WT);
  __bf16* Wsct  = carve(WT);
  __bf16* Wsht  = carve(WT);
  __bf16* Qb    = carve(ACT);
  __bf16* Krb   = carve(ACT);
  __bf16* Vrb   = carve(ACT);
  __bf16* Scb   = carve(ACT);
  __bf16* Shb   = carve(ACT);
  __bf16* Kfb   = carve(ACT);
  __bf16* Vtb   = carve(ACT);   // [B,H,HD,S]
  __bf16* Obuf  = carve(ACT);

  const int actN = MROWS * D_;  // 4,194,304

  // 1) casts
  cast_bf16_kernel<<<actN / 1024, 256, 0, stream>>>(mod1, mod1b, actN);
  cast_bf16_kernel<<<actN / 1024, 256, 0, stream>>>(mod2, mod2b, actN);
  castT_bf16_kernel<<<(D_ * D_) / 256, 256, 0, stream>>>(Wq, Wqt);
  castT_bf16_kernel<<<(D_ * D_) / 256, 256, 0, stream>>>(Wk, Wkt);
  castT_bf16_kernel<<<(D_ * D_) / 256, 256, 0, stream>>>(Wv, Wvt);
  castT_bf16_kernel<<<(D_ * D_) / 256, 256, 0, stream>>>(Wo, Wot);
  castT_bf16_kernel<<<(D_ * D_) / 256, 256, 0, stream>>>(Wsc, Wsct);
  castT_bf16_kernel<<<(D_ * D_) / 256, 256, 0, stream>>>(Wsh, Wsht);

  // 2) projections (Q carries 1/sqrt(64))
  dim3 gg(D_ / BN, MROWS / BM);  // (4, 64)
  gemm_bf16_kernel<false><<<gg, 256, 0, stream>>>(mod1b, Wqt, bq, Qb,
                                                  MROWS, D_, D_, 0.125f);
  gemm_bf16_kernel<false><<<gg, 256, 0, stream>>>(mod2b, Wkt, bk, Krb,
                                                  MROWS, D_, D_, 1.0f);
  gemm_bf16_kernel<false><<<gg, 256, 0, stream>>>(mod2b, Wvt, bv, Vrb,
                                                  MROWS, D_, D_, 1.0f);
  gemm_bf16_kernel<false><<<gg, 256, 0, stream>>>(mod2b, Wsct, bsc, Scb,
                                                  MROWS, D_, D_, 1.0f);
  gemm_bf16_kernel<false><<<gg, 256, 0, stream>>>(mod2b, Wsht, bsh, Shb,
                                                  MROWS, D_, D_, 1.0f);

  // 3) K' = K*s+sh, V' = V*s+sh (and per-head transposed V')
  fuse_kv_kernel<<<actN / 256, 256, 0, stream>>>(Krb, Vrb, Scb, Shb, Kfb, Vtb);

  // 4) flash attention
  attn_kernel<<<dim3(S_ / 64, H_, B_), 128, 0, stream>>>(Qb, Kfb, Vtb, Obuf);

  // 5) output projection -> f32
  gemm_bf16_kernel<true><<<gg, 256, 0, stream>>>(Obuf, Wot, bo, out,
                                                 MROWS, D_, D_, 1.0f);
}
